// InjectionLayer_91130616086689
// MI455X (gfx1250) — matrix operand order repaired and checked by reading, not compiled, not run on hardware
//
#include <hip/hip_runtime.h>
#include <hip/hip_bf16.h>
#include <math.h>

// ---------------------------------------------------------------------------
// InjectionLayer for MI455X (gfx1250, wave32).
//
// GEMMs: v_wmma_f32_16x16x32_bf16 with Ootomo hi/lo fp32->bf16 split
// (3 WMMAs per 32-wide K chunk, fp32 accumulators -> ~fp32 accuracy).
// All bf16 conversion is hoisted OUT of the WMMA hot loop:
//   * weights are pre-packed once per launch into fragment-major bf16 hi/lo
//     buffers in d_ws (B fragment = one contiguous 32B load per lane),
//   * activation tiles are converted once per block into fragment-major
//     bf16 hi/lo LDS (A fragment = one contiguous 32B ds load per lane).
// Graph phase (gather + f32 atomic scatter) is L2-resident (51MB arrays
// << 192MB L2); scatter prefetches the next edge's source row.
// ---------------------------------------------------------------------------

typedef __attribute__((ext_vector_type(16))) __bf16 v16bf;
typedef __attribute__((ext_vector_type(8)))  float  v8f;

#define HID 128
#define EVD 64

// packed-weight layout: per matrix, per K-chunk c (32 rows), per 16-col group g,
// 32 lanes x 16 bf16 contiguous:  off = ((c*8 + g)*32 + lane)*16 + i
// Consumer (B fragment, 32x16 KxN, ISA layout): lane = (n&15) + 16*bit3(koff),
// element i = (koff&7) + 8*bit4(koff).
#define PROJ_OFF 0            // 64x128  -> 2 chunks * 8 groups * 512
#define GCN_OFF  8192         // 128x128 -> 4 chunks
#define GATE_OFF 24576        // 256x128 -> 8 chunks
#define FUSE_OFF 57344        // 256x128 -> 8 chunks
#define WTOT     90112

__device__ __forceinline__ int lane_id() { return (int)(threadIdx.x & 31u); }

// fp32 -> bf16 hi/lo split (RTNE both): a ~= hi + lo
__device__ __forceinline__ void bf16_split(float a, __bf16& hi, __bf16& lo) {
    __bf16 h = (__bf16)a;
    hi = h;
    lo = (__bf16)(a - (float)h);
}

// A-fragment scatter address for element (m, koff) of a 16x32 bf16 chunk.
// ISA 16-bit A layout: lane = m + 16*bit3(koff); element i = (koff&7)|8*bit4(koff)
__device__ __forceinline__ int a_frag_addr(int chunk, int m, int koff) {
    const int ln = m + 16 * ((koff >> 3) & 1);
    const int i  = (koff & 7) + 8 * ((koff >> 4) & 1);
    return (chunk * 32 + ln) * 16 + i;
}

__device__ __forceinline__ v16bf ld16(const __bf16* p) {
    return *(const v16bf*)p;
}

// 3-term compensated bf16 WMMA: acc += A_hi*B_hi + A_hi*B_lo + A_lo*B_hi
__device__ __forceinline__ v8f wmma3(v8f acc, v16bf ahi, v16bf alo,
                                     v16bf bhi, v16bf blo) {
    acc = __builtin_amdgcn_wmma_f32_16x16x32_bf16(false, alo, false, bhi,
                                                  (short)0, acc, false, false);
    acc = __builtin_amdgcn_wmma_f32_16x16x32_bf16(false, ahi, false, blo,
                                                  (short)0, acc, false, false);
    acc = __builtin_amdgcn_wmma_f32_16x16x32_bf16(false, ahi, false, bhi,
                                                  (short)0, acc, false, false);
    return acc;
}

// ---------------------------------------------------------------------------
// Weight pre-pack: row-major f32 W[K x 128] -> fragment-major bf16 hi/lo.
// ---------------------------------------------------------------------------
__global__ void k_wconv(const float* __restrict__ W, int K,
                        __bf16* __restrict__ whi, __bf16* __restrict__ wlo)
{
    const int idx = blockIdx.x * blockDim.x + threadIdx.x;
    if (idx >= K * HID) return;
    const int k = idx >> 7, nn = idx & 127;
    const int c = k >> 5, koff = k & 31;
    const int g = nn >> 4;
    const int lane = (nn & 15) + 16 * ((koff >> 3) & 1);
    const int i = (koff & 7) + 8 * ((koff >> 4) & 1);
    const int dst = ((c * 8 + g) * 32 + lane) * 16 + i;
    __bf16 h, lo;
    bf16_split(W[idx], h, lo);
    whi[dst] = h;
    wlo[dst] = lo;
}

// ---------------------------------------------------------------------------
// Kernel 1: event_feat = relu(ev @ W_proj + b_proj); x = event_feat @ W_gcn
// One block = 16 nodes, 8 waves; wave w owns output columns [16w,16w+16).
// ---------------------------------------------------------------------------
__global__ __launch_bounds__(256) void k_proj_gcnlin(
    const float* __restrict__ ev, const float* __restrict__ bp,
    const __bf16* __restrict__ whi, const __bf16* __restrict__ wlo,
    float* __restrict__ x, int n)
{
    __shared__ __bf16 aHi[4 * 512];   // up to 4 K-chunks, fragment-major
    __shared__ __bf16 aLo[4 * 512];
    const int row0 = blockIdx.x * 16;

    // stage event tile [16x64] directly as hi/lo fragments (chunks 0..1)
    for (int t = threadIdx.x; t < 16 * EVD; t += 256) {
        const int r = t >> 6, k = t & 63;
        int gr = row0 + r; gr = gr < n ? gr : n - 1;
        const int ad = a_frag_addr(k >> 5, r, k & 31);
        __bf16 h, lo;
        bf16_split(ev[(size_t)gr * EVD + k], h, lo);
        aHi[ad] = h; aLo[ad] = lo;
    }
    __syncthreads();

    const int g = (int)(threadIdx.x >> 5);   // wave = column group
    const int l = lane_id();
    const int m = l & 15, half = l >> 4;
    const int col = g * 16 + m;
    const int lb = l * 16;                   // lane base inside a fragment

    // GEMM1: [16x64] @ [64x128]
    v8f acc = {};
#pragma unroll
    for (int c = 0; c < 2; ++c) {
        const v16bf ah = ld16(aHi + c * 512 + lb);
        const v16bf al = ld16(aLo + c * 512 + lb);
        const v16bf bh = ld16(whi + PROJ_OFF + (c * 8 + g) * 512 + lb);
        const v16bf bl = ld16(wlo + PROJ_OFF + (c * 8 + g) * 512 + lb);
        acc = wmma3(acc, ah, al, bh, bl);
    }
    __syncthreads();   // everyone done reading event fragments

    // event_feat = relu(acc + bias) -> A fragments for GEMM2 (chunks 0..3)
    const float bias = bp[col];
    const int cc = col >> 5, koff = col & 31;
#pragma unroll
    for (int r = 0; r < 8; ++r) {
        const int mm = r + half * 8;
        const float v = acc[r] + bias;
        const int ad = a_frag_addr(cc, mm, koff);
        __bf16 h, lo;
        bf16_split(v > 0.f ? v : 0.f, h, lo);
        aHi[ad] = h; aLo[ad] = lo;
    }
    __syncthreads();

    // GEMM2: [16x128] @ [128x128]  (bias added after aggregation)
    v8f acc2 = {};
#pragma unroll
    for (int c = 0; c < 4; ++c) {
        const v16bf ah = ld16(aHi + c * 512 + lb);
        const v16bf al = ld16(aLo + c * 512 + lb);
        const v16bf bh = ld16(whi + GCN_OFF + (c * 8 + g) * 512 + lb);
        const v16bf bl = ld16(wlo + GCN_OFF + (c * 8 + g) * 512 + lb);
        acc2 = wmma3(acc2, ah, al, bh, bl);
    }
#pragma unroll
    for (int r = 0; r < 8; ++r) {
        const int mm = r + half * 8;
        if (row0 + mm < n)
            x[(size_t)(row0 + mm) * HID + col] = acc2[r];
    }
}

// ---------------------------------------------------------------------------
// Degree + normalization
// ---------------------------------------------------------------------------
__global__ void k_deg(const int* __restrict__ dst, float* __restrict__ deg, int E)
{
    const int e = blockIdx.x * blockDim.x + threadIdx.x;
    if (e < E) atomicAdd(&deg[dst[e]], 1.0f);
}

__global__ void k_dis(float* __restrict__ deg_dis, int n)
{
    const int i = blockIdx.x * blockDim.x + threadIdx.x;
    if (i < n) deg_dis[i] = rsqrtf(deg_dis[i] + 1.0f);  // +1 = self loop => deg>0
}

// ---------------------------------------------------------------------------
// Scatter: one wave per edge; lane j handles 4 channels via float4 gather,
// 4x global_atomic_add_f32 into agg[dst] (L2-resident). Prefetch next row.
// ---------------------------------------------------------------------------
__global__ __launch_bounds__(256) void k_scatter(
    const int* __restrict__ src, const int* __restrict__ dstv,
    const float* __restrict__ dis, const float* __restrict__ x,
    float* __restrict__ agg, int E)
{
    const int gw = (int)((blockIdx.x * blockDim.x + threadIdx.x) >> 5);
    const int nw = (int)((gridDim.x * blockDim.x) >> 5);
    const int l = lane_id();
    for (int e = gw; e < E; e += nw) {
        const int e2 = e + nw;
        if (e2 < E)   // global_prefetch_b8 of the next source row (L2 hint)
            __builtin_prefetch(x + (size_t)src[e2] * HID + l * 4, 0, 3);
        const int s = src[e], d = dstv[e];
        const float w = dis[s] * dis[d];
        const float4 v = ((const float4*)(x + (size_t)s * HID))[l];
        float* ad = agg + (size_t)d * HID + l * 4;
        atomicAdd(ad + 0, v.x * w);
        atomicAdd(ad + 1, v.y * w);
        atomicAdd(ad + 2, v.z * w);
        atomicAdd(ad + 3, v.w * w);
    }
}

// ---------------------------------------------------------------------------
// diffused = relu(agg + x*dis^2 + b_gcn), in place over agg. float4 per thread.
// ---------------------------------------------------------------------------
__global__ void k_diffuse(const float* __restrict__ x, const float* __restrict__ dis,
                          const float* __restrict__ bg, float* __restrict__ agg, int n)
{
    const int idx = blockIdx.x * blockDim.x + threadIdx.x;
    if (idx >= n * (HID / 4)) return;
    const int node = idx >> 5;
    const int q = idx & 31;
    const float di = dis[node];
    const float sw = di * di;
    const float4 xv = ((const float4*)(x + (size_t)node * HID))[q];
    float4* av = (float4*)(agg + (size_t)node * HID);
    float4 ag = av[q];
    const float4 b = ((const float4*)bg)[q];
    float4 r;
    r.x = fmaxf(ag.x + xv.x * sw + b.x, 0.f);
    r.y = fmaxf(ag.y + xv.y * sw + b.y, 0.f);
    r.z = fmaxf(ag.z + xv.z * sw + b.z, 0.f);
    r.w = fmaxf(ag.w + xv.w * sw + b.w, 0.f);
    av[q] = r;
}

// ---------------------------------------------------------------------------
// Kernel 6: gate = sigmoid([base|diff] @ W_gate + b); corrected = lerp;
//           out = relu([base|corrected] @ W_fuse + b)
// ---------------------------------------------------------------------------
__global__ __launch_bounds__(256) void k_gate_fuse(
    const float* __restrict__ base, const float* __restrict__ diff,
    const __bf16* __restrict__ whi, const __bf16* __restrict__ wlo,
    const float* __restrict__ bgate, const float* __restrict__ bfuse,
    float* __restrict__ out, int n)
{
    __shared__ __bf16 bHi[4 * 512];   // base tile fragments
    __shared__ __bf16 bLo[4 * 512];
    __shared__ __bf16 cHi[4 * 512];   // diffused, later corrected
    __shared__ __bf16 cLo[4 * 512];
    const int row0 = blockIdx.x * 16;

    for (int t = threadIdx.x; t < 16 * HID; t += 256) {
        const int r = t >> 7, k = t & 127;
        int gr = row0 + r; gr = gr < n ? gr : n - 1;
        const int ad = a_frag_addr(k >> 5, r, k & 31);
        __bf16 h, lo;
        bf16_split(base[(size_t)gr * HID + k], h, lo);
        bHi[ad] = h; bLo[ad] = lo;
        bf16_split(diff[(size_t)gr * HID + k], h, lo);
        cHi[ad] = h; cLo[ad] = lo;
    }
    __syncthreads();

    const int g = (int)(threadIdx.x >> 5);
    const int l = lane_id();
    const int m = l & 15, half = l >> 4;
    const int col = g * 16 + m;
    const int lb = l * 16;

    // gate GEMM: K=256 = base rows (chunks 0..3) + diff rows (chunks 4..7)
    v8f acc = {};
#pragma unroll
    for (int c = 0; c < 4; ++c) {
        v16bf ah = ld16(bHi + c * 512 + lb);
        v16bf al = ld16(bLo + c * 512 + lb);
        v16bf bh = ld16(whi + GATE_OFF + (c * 8 + g) * 512 + lb);
        v16bf bl = ld16(wlo + GATE_OFF + (c * 8 + g) * 512 + lb);
        acc = wmma3(acc, ah, al, bh, bl);
        ah = ld16(cHi + c * 512 + lb);
        al = ld16(cLo + c * 512 + lb);
        bh = ld16(whi + GATE_OFF + ((c + 4) * 8 + g) * 512 + lb);
        bl = ld16(wlo + GATE_OFF + ((c + 4) * 8 + g) * 512 + lb);
        acc = wmma3(acc, ah, al, bh, bl);
    }

    const float gb = bgate[col];
    const int cc = col >> 5, koff = col & 31;
    __syncthreads();   // all waves finished reading diff fragments
#pragma unroll
    for (int r = 0; r < 8; ++r) {
        const int mm = r + half * 8;
        const int ad = a_frag_addr(cc, mm, koff);
        const float gv = 1.f / (1.f + expf(-(acc[r] + gb)));
        const float bv = (float)bHi[ad] + (float)bLo[ad];
        const float dv = (float)cHi[ad] + (float)cLo[ad];
        __bf16 h, lo;
        bf16_split(bv * (1.f - gv) + dv * gv, h, lo);   // corrected
        cHi[ad] = h; cLo[ad] = lo;
    }
    __syncthreads();

    // fuse GEMM: K=256 = base rows (chunks 0..3) + corrected rows (chunks 4..7)
    v8f acc2 = {};
#pragma unroll
    for (int c = 0; c < 4; ++c) {
        v16bf ah = ld16(bHi + c * 512 + lb);
        v16bf al = ld16(bLo + c * 512 + lb);
        v16bf bh = ld16(whi + FUSE_OFF + (c * 8 + g) * 512 + lb);
        v16bf bl = ld16(wlo + FUSE_OFF + (c * 8 + g) * 512 + lb);
        acc2 = wmma3(acc2, ah, al, bh, bl);
        ah = ld16(cHi + c * 512 + lb);
        al = ld16(cLo + c * 512 + lb);
        bh = ld16(whi + FUSE_OFF + ((c + 4) * 8 + g) * 512 + lb);
        bl = ld16(wlo + FUSE_OFF + ((c + 4) * 8 + g) * 512 + lb);
        acc2 = wmma3(acc2, ah, al, bh, bl);
    }
    const float fb = bfuse[col];
#pragma unroll
    for (int r = 0; r < 8; ++r) {
        const int mm = r + half * 8;
        if (row0 + mm < n)
            out[(size_t)(row0 + mm) * HID + col] = fmaxf(acc2[r] + fb, 0.f);
    }
}

// ---------------------------------------------------------------------------
extern "C" void kernel_launch(void* const* d_in, const int* in_sizes, int n_in,
                              void* d_out, int out_size, void* d_ws, size_t ws_size,
                              hipStream_t stream)
{
    const float* base  = (const float*)d_in[0];   // [N,128]
    const float* ev    = (const float*)d_in[1];   // [N,64]
    const int*   eidx  = (const int*)d_in[2];     // [2,E] (int32)
    const float* Wp    = (const float*)d_in[3];   // [64,128]
    const float* bp    = (const float*)d_in[4];
    const float* Wg    = (const float*)d_in[5];   // [128,128]
    const float* bg    = (const float*)d_in[6];
    const float* Wgate = (const float*)d_in[7];   // [256,128]
    const float* bgate = (const float*)d_in[8];
    const float* Wf    = (const float*)d_in[9];   // [256,128]
    const float* bfu   = (const float*)d_in[10];
    float* out = (float*)d_out;

    const int n = in_sizes[0] / HID;       // 100000
    const int E = in_sizes[2] / 2;         // 1600000
    const int* srcv = eidx;
    const int* dstv = eidx + E;

    // workspace: x[N*128] | agg[N*128] | deg->dis[N] | whi[WTOT] | wlo[WTOT]
    float* x   = (float*)d_ws;
    float* agg = x + (size_t)n * HID;
    float* dis = agg + (size_t)n * HID;
    size_t foff = (size_t)2 * n * HID + n;
    foff = (foff + 7) & ~(size_t)7;                 // 32B align for b128 loads
    __bf16* whi = (__bf16*)((float*)d_ws + foff);
    __bf16* wlo = whi + WTOT;

    // zero agg + deg every call (deterministic, capture-safe)
    (void)hipMemsetAsync(agg, 0, ((size_t)n * HID + n) * sizeof(float), stream);

    // pre-pack weights into fragment-major bf16 hi/lo
    k_wconv<<<(64  * HID + 255) / 256, 256, 0, stream>>>(Wp,    64,  whi + PROJ_OFF, wlo + PROJ_OFF);
    k_wconv<<<(128 * HID + 255) / 256, 256, 0, stream>>>(Wg,    128, whi + GCN_OFF,  wlo + GCN_OFF);
    k_wconv<<<(256 * HID + 255) / 256, 256, 0, stream>>>(Wgate, 256, whi + GATE_OFF, wlo + GATE_OFF);
    k_wconv<<<(256 * HID + 255) / 256, 256, 0, stream>>>(Wf,    256, whi + FUSE_OFF, wlo + FUSE_OFF);

    const int tiles = (n + 15) / 16;
    k_proj_gcnlin<<<tiles, 256, 0, stream>>>(ev, bp, whi, wlo, x, n);
    k_deg<<<(E + 255) / 256, 256, 0, stream>>>(dstv, dis, E);
    k_dis<<<(n + 255) / 256, 256, 0, stream>>>(dis, n);
    k_scatter<<<2048, 256, 0, stream>>>(srcv, dstv, dis, x, agg, E);
    k_diffuse<<<(n * (HID / 4) + 255) / 256, 256, 0, stream>>>(x, dis, bg, agg, n);
    k_gate_fuse<<<tiles, 256, 0, stream>>>(base, agg, whi, wlo, bgate, bfu, out, n);
}
